// SparseWeights_84344567759316
// MI455X (gfx1250) — matrix operand order repaired and checked by reading, not compiled, not run on hardware
//
#include <hip/hip_runtime.h>

typedef __attribute__((ext_vector_type(16))) _Float16 v16h;
typedef __attribute__((ext_vector_type(8)))  _Float16 v8h;
typedef __attribute__((ext_vector_type(4)))  _Float16 v4h;
typedef __attribute__((ext_vector_type(8)))  float    v8f;
typedef __attribute__((ext_vector_type(4)))  float    v4f;
typedef __attribute__((ext_vector_type(4)))  unsigned int u32x4;
typedef __attribute__((ext_vector_type(8)))  int      i32x8;
typedef __attribute__((ext_vector_type(4)))  int      i32x4;

#define T_TOK 8192
#define N_IN  4096
#define M_OUT 4096
#define BK    64
#define LDSW  72   // halfs per LDS row (64 data + 8 pad); 144B = 128B data + 16B pad

#if __has_builtin(__builtin_amdgcn_tensor_load_to_lds) && \
    __has_builtin(__builtin_amdgcn_s_wait_tensorcnt)
#define USE_TDM 1
#else
#define USE_TDM 0
#endif

// ---------------- Kernel 1: W' = f16(W_dense), elementwise convert ----------------
__global__ __launch_bounds__(256) void wconv_kernel(const float* __restrict__ w,
                                                    _Float16* __restrict__ wh) {
    size_t i = ((size_t)blockIdx.x * blockDim.x + threadIdx.x) * 4;
    v4f f = *(const v4f*)(w + i);
    v4h h;
    h[0] = (_Float16)f[0]; h[1] = (_Float16)f[1];
    h[2] = (_Float16)f[2]; h[3] = (_Float16)f[3];
    *(v4h*)(wh + i) = h;
}

// ---------------- Kernel 2: scatter sparse values into W' (unique (m,n) pairs) ----
__global__ __launch_bounds__(256) void wscatter_kernel(const float* __restrict__ dense,
                                                       const float* __restrict__ vals,
                                                       const int* __restrict__ rows,
                                                       const int* __restrict__ cols,
                                                       _Float16* __restrict__ wh,
                                                       int nnz) {
    int i = blockIdx.x * blockDim.x + threadIdx.x;
    if (i < nnz) {
        size_t off = (size_t)rows[i] * N_IN + cols[i];
        wh[off] = (_Float16)(dense[off] + vals[i]);
    }
}

#if USE_TDM
// Issue one TDM transfer: 128 rows x 64 f16 tile of W' (row stride N_IN halfs) into
// LDS at lds_off, padded +16B every 128B so rows land at stride LDSW (144B).
__device__ __forceinline__ void issue_w_tdm(const _Float16* gsrc, unsigned lds_off) {
    unsigned long long ga = (unsigned long long)(uintptr_t)gsrc;
    u32x4 g0;
    g0[0] = 1u;                                        // count=1 (valid), user mode
    g0[1] = lds_off;                                   // lds_addr (bytes)
    g0[2] = (unsigned)(ga & 0xFFFFFFFFu);              // global_addr[31:0]
    g0[3] = (unsigned)((ga >> 32) & 0x1FFFFFFu)        // global_addr[56:32]
          | (2u << 30);                                // type=2 ("image")
    i32x8 g1;
    g1[0] = (int)((1u << 16)     // data_size = 2 bytes
                | (1u << 20)     // pad_enable
                | (4u << 22)     // pad_interval: 32 DWORDs = 128B (one tile row)
                | (3u << 25));   // pad_amount: 4 DWORDs = 16B
    g1[1] = (int)(0x1000u << 16);   // tensor_dim0 = 4096 (low16)
    g1[2] = (int)(0x1000u << 16);   // tensor_dim0 hi=0 | tensor_dim1 = 4096 (low16)
    g1[3] = (int)(64u << 16);       // tensor_dim1 hi=0 | tile_dim0 = 64
    g1[4] = 128;                    // tile_dim1 = 128 | tile_dim2 = 0
    g1[5] = 4096;                   // tensor_dim0_stride[31:0] = 4096 elements
    g1[6] = 0;                      // dim0_stride hi | dim1_stride lo (2D: unused)
    g1[7] = 0;
    i32x4 z4 = {0, 0, 0, 0};        // groups 2/3: 2D tensor -> zeros
    i32x8 z8 = {0, 0, 0, 0, 0, 0, 0, 0};
    __builtin_amdgcn_tensor_load_to_lds(g0, g1, z4, z4, z8, 0);
}
__device__ __forceinline__ unsigned lds_byte_off(const void* p) {
    return (unsigned)(unsigned long long)(uintptr_t)p;
}
#endif

// ---------------- Kernel 3: y = x @ W'^T + b via v_wmma_f32_16x16x32_f16 ----------
// Block: 256 threads = 8 waves. Tile: 128 tokens x 128 m, K-step 64, double-buffered.
// W' tiles streamed by the Tensor Data Mover (wave 0, TENSORcnt); x tiles converted
// f32->f16 by all waves. 16 wmma per K-iteration.
__global__ __launch_bounds__(256) void gemm_kernel(const float* __restrict__ x,
                                                   const _Float16* __restrict__ wh,
                                                   const float* __restrict__ bias,
                                                   float* __restrict__ out) {
    __shared__ _Float16 xs[2][128 * LDSW];
    __shared__ _Float16 wsm[2][128 * LDSW];

    const int tid   = threadIdx.x;
    const int lane  = tid & 31;
    const int wave  = tid >> 5;
    const int waveT = wave & 3;     // 0..3 token direction
    const int waveM = wave >> 2;    // 0..1 m direction
    const int lhalf = lane & 15;
    const int hsel  = lane >> 4;

    const int mBlock = blockIdx.x * 128;
    const int tBlock = blockIdx.y * 128;

    v8f acc[2][4];
#pragma unroll
    for (int i = 0; i < 2; ++i)
#pragma unroll
        for (int j = 0; j < 4; ++j)
#pragma unroll
            for (int e = 0; e < 8; ++e) acc[i][j][e] = 0.0f;

    // ---- x tile fill: 128 rows x 64 f32 -> f16 LDS (8 float4 chunks per thread) --
    auto fillX = [&](int sel, int kt) {
#pragma unroll
        for (int q = 0; q < 8; ++q) {
            int chunk = q * 256 + tid;
            int row   = chunk >> 4;
            int kc    = (chunk & 15) * 4;
            v4f f = *(const v4f*)(x + (size_t)(tBlock + row) * N_IN + kt + kc);
            v4h h;
            h[0] = (_Float16)f[0]; h[1] = (_Float16)f[1];
            h[2] = (_Float16)f[2]; h[3] = (_Float16)f[3];
            *(v4h*)(&xs[sel][row * LDSW + kc]) = h;
        }
    };
#if !USE_TDM
    // ---- fallback W' tile fill (manual) ------------------------------------------
    auto fillW = [&](int sel, int kt) {
#pragma unroll
        for (int q = 0; q < 4; ++q) {
            int chunk = q * 256 + tid;
            int row   = chunk >> 3;
            int kc    = (chunk & 7) * 8;
            v8h h = *(const v8h*)(wh + (size_t)(mBlock + row) * N_IN + kt + kc);
            *(v8h*)(&wsm[sel][row * LDSW + kc]) = h;
        }
    };
#endif

    // ---- compute one BK=64 tile from buffer `sel` (2 k-steps, 16 wmma) -----------
    auto compute = [&](int sel) {
#pragma unroll
        for (int ks = 0; ks < BK; ks += 32) {
            v16h a[2];
#pragma unroll
            for (int i = 0; i < 2; ++i) {
                const _Float16* p =
                    &xs[sel][(waveT * 32 + i * 16 + lhalf) * LDSW + ks + hsel * 8];
                v8h lo = *(const v8h*)p;
                v8h hi = *(const v8h*)(p + 16);
#pragma unroll
                for (int e = 0; e < 8; ++e) { a[i][e] = lo[e]; a[i][8 + e] = hi[e]; }
            }
            v16h b[4];
#pragma unroll
            for (int j = 0; j < 4; ++j) {
                const _Float16* p =
                    &wsm[sel][(waveM * 64 + j * 16 + lhalf) * LDSW + ks + hsel * 8];
                v8h lo = *(const v8h*)p;
                v8h hi = *(const v8h*)(p + 16);
#pragma unroll
                for (int e = 0; e < 8; ++e) { b[j][e] = lo[e]; b[j][8 + e] = hi[e]; }
            }
#pragma unroll
            for (int i = 0; i < 2; ++i)
#pragma unroll
                for (int j = 0; j < 4; ++j)
                    acc[i][j] = __builtin_amdgcn_wmma_f32_16x16x32_f16(
                        false, a[i], false, b[j], (short)0, acc[i][j], false, false);
        }
    };

    // ---- prologue: fill buffer 0 -------------------------------------------------
    fillX(0, 0);
#if USE_TDM
    if (wave == 0) {
        issue_w_tdm(wh + (size_t)mBlock * N_IN, lds_byte_off(&wsm[0][0]));
        __builtin_amdgcn_s_wait_tensorcnt(0);
    }
#else
    fillW(0, 0);
#endif
    __syncthreads();

    // ---- main loop: double-buffered, TDM + x-fill overlap compute ----------------
    int cur = 0;
    for (int kt = 0; kt < N_IN; kt += BK) {
        const bool more = (kt + BK) < N_IN;
#if USE_TDM
        if (wave == 0 && more)
            issue_w_tdm(wh + (size_t)mBlock * N_IN + kt + BK,
                        lds_byte_off(&wsm[cur ^ 1][0]));
#endif
        if (more) fillX(cur ^ 1, kt + BK);
#if !USE_TDM
        if (more) fillW(cur ^ 1, kt + BK);
#endif
        compute(cur);
#if USE_TDM
        if (wave == 0 && more) __builtin_amdgcn_s_wait_tensorcnt(0);
#endif
        __syncthreads();
        cur ^= 1;
    }

    // ---- epilogue: row t = vgpr + hsel*8, col m = lhalf; add bias ----------------
#pragma unroll
    for (int i = 0; i < 2; ++i) {
        int t0 = tBlock + waveT * 32 + i * 16 + hsel * 8;
#pragma unroll
        for (int j = 0; j < 4; ++j) {
            int m = mBlock + waveM * 64 + j * 16 + lhalf;
            float bm = bias[m];
#pragma unroll
            for (int v = 0; v < 8; ++v)
                out[(size_t)(t0 + v) * M_OUT + m] = acc[i][j][v] + bm;
        }
    }
}

extern "C" void kernel_launch(void* const* d_in, const int* in_sizes, int n_in,
                              void* d_out, int out_size, void* d_ws, size_t ws_size,
                              hipStream_t stream) {
    const float* x      = (const float*)d_in[0];   // [T, N]
    const float* wdense = (const float*)d_in[1];   // [M, N]
    const float* bias   = (const float*)d_in[2];   // [M]
    const float* svals  = (const float*)d_in[3];   // [nnz]
    const int*   rows   = (const int*)d_in[4];     // [nnz]
    const int*   cols   = (const int*)d_in[5];     // [nnz]
    float*       out    = (float*)d_out;           // [T, M]
    _Float16*    wh     = (_Float16*)d_ws;         // [M, N] f16 (33.5 MB scratch)

    const int nnz = in_sizes[3];

    // 1) convert frozen dense weight to f16
    {
        size_t elems = (size_t)M_OUT * N_IN;
        int blocks = (int)(elems / 4 / 256);
        wconv_kernel<<<blocks, 256, 0, stream>>>(wdense, wh);
    }
    // 2) scatter-add sparse values (unique (row,col) -> race-free overwrite)
    {
        int blocks = (nnz + 255) / 256;
        wscatter_kernel<<<blocks, 256, 0, stream>>>(wdense, svals, rows, cols, wh, nnz);
    }
    // 3) WMMA GEMM with fused bias (TDM-fed, double-buffered LDS pipeline)
    {
        dim3 grid(M_OUT / 128, T_TOK / 128);       // (32, 64)
        gemm_kernel<<<grid, 256, 0, stream>>>(x, wh, bias, out);
    }
}
